// EdgeDecoder_24257975287899
// MI455X (gfx1250) — matrix operand order repaired and checked by reading, not compiled, not run on hardware
//
#include <hip/hip_runtime.h>

typedef _Float16 f16;
typedef __attribute__((ext_vector_type(16))) _Float16 v16h;
typedef __attribute__((ext_vector_type(8)))  _Float16 v8h;
typedef __attribute__((ext_vector_type(4)))  _Float16 v4h;
typedef __attribute__((ext_vector_type(8)))  float    v8f;

#define HDIM 128
#define WAVES_PER_BLOCK 4
#define THREADS (WAVES_PER_BLOCK * 32)
#define WPACK_HALVES (HDIM * HDIM)      // 16384 halves per packed weight matrix
#define TILE_HALVES 512                 // one 16x32 A tile: 32 lanes * 16 halves
#define M_PER_WAVE 32                   // 32 edges per wave (two 16-edge N-tiles)

// LDS byte layout:
//   [0,      32768) W1^T packed f16 (A-operand layout)
//   [32768,  65536) W2^T packed f16 (A-operand layout)
//   [65536,  66048) b1 (f32 x128)
//   [66048,  66560) b2 (f32 x128)
//   [66560,  67072) W3 (f32 x128)
//   [67072, ...)    per-wave activation buffers: 4 waves * (Abuf 8KB + Hbuf 8KB)
#define WAVE_BUF_HALVES (2 * M_PER_WAVE * HDIM)
#define LDS_BYTES (2 * WPACK_HALVES * 2 + 3 * HDIM * 4 + \
                   WAVES_PER_BLOCK * WAVE_BUF_HALVES * 2)

// B-operand fragment (32x16, K x edges) read from a row-major [edge][128] f16
// buffer. K-contiguous order, as in W packing: lanes 0-15 hold K=kt*32+0..15, lanes 16-31
// hold K=kt*32+16..31; halves K-contiguous per lane -> one 32B contiguous read.
__device__ inline v16h load_B_act(const f16* buf, int lane, int kt) {
    const f16* p = buf + (lane & 15) * HDIM + kt * 32 + ((lane & 16) ? 16 : 0);
    return *(const v16h*)p;
}

__global__ __launch_bounds__(THREADS)
void EdgeDecoder_wmma_kernel(const float* __restrict__ x,
                             const long long* __restrict__ src,
                             const long long* __restrict__ dst,
                             const float* __restrict__ W1, const float* __restrict__ b1,
                             const float* __restrict__ W2, const float* __restrict__ b2,
                             const float* __restrict__ W3, const float* __restrict__ b3,
                             float* __restrict__ out,
                             int nEdges, int nTiles, int totalWaves) {
    extern __shared__ char smem[];
    f16*   W1a  = (f16*)smem;                 // W1^T in A-operand layout
    f16*   W2a  = W1a + WPACK_HALVES;         // W2^T in A-operand layout
    float* b1s  = (float*)(W2a + WPACK_HALVES);
    float* b2s  = b1s + HDIM;
    float* W3s  = b2s + HDIM;
    f16*   wbuf = (f16*)(W3s + HDIM);

    const int tid = threadIdx.x;

    // ---- One-time per block: pack W^T into a K-contiguous 16-bit A layout (same order as B).
    // Tile T = mt*4 + kt covers channels [mt*16,mt*16+16) x K [kt*32,kt*32+32).
    // Lane L, half h:  row (channel) = mt*16 + (L&15)
    //                  K = kt*32 + (L>=16 ? 16 : 0) + h
    // A[m][k] of W^T is W[k][m]  ->  source element W[K][channel].
    for (int i = tid; i < WPACK_HALVES; i += THREADS) {
        const int T = i >> 9, rem = i & 511;
        const int L = rem >> 4, h = rem & 15;
        const int mt = T >> 2, kt = T & 3;
        const int col  = mt * 16 + (L & 15);
        const int koff = (h & 7) + ((h & 8) << 1);          // 0..7, 16..23
        const int k    = kt * 32 + ((L & 16) ? 16 : 0) + h;
        W1a[i] = (f16)W1[k * HDIM + col];
        W2a[i] = (f16)W2[k * HDIM + col];
    }
    for (int i = tid; i < HDIM; i += THREADS) {
        b1s[i] = b1[i]; b2s[i] = b2[i]; W3s[i] = W3[i];
    }
    __syncthreads();

    const int lane  = tid & 31;
    const int wave  = tid >> 5;
    const int gwave = blockIdx.x * WAVES_PER_BLOCK + wave;
    f16* Abuf = wbuf + wave * WAVE_BUF_HALVES;        // [32 edges][128] f16
    f16* Hbuf = Abuf + M_PER_WAVE * HDIM;             // [32 edges][128] f16

    const int   n2    = lane & 15;           // edge column within a 16-wide N tile
    const int   mBase = (lane & 16) ? 8 : 0; // C layout: channel rows r / r+8
    const float b3v   = b3[0];

    for (int t = gwave; t < nTiles; t += totalWaves) {
        const int e0 = t * M_PER_WAVE;

        // Prefetch next tile's edge indices into L2 (global_prefetch_b8).
        if (t + totalWaves < nTiles) {
            __builtin_prefetch(src + (size_t)(t + totalWaves) * M_PER_WAVE, 0, 0);
            __builtin_prefetch(dst + (size_t)(t + totalWaves) * M_PER_WAVE, 0, 0);
        }

        int ei = e0 + lane; if (ei >= nEdges) ei = nEdges - 1;
        const int sIdx = (int)src[ei];
        const int dIdx = (int)dst[ei];

        // ---- Build A = x[src] * x[dst] (32 edges x 128, f16, row-major).
#pragma unroll
        for (int m = 0; m < M_PER_WAVE; ++m) {
            const int s = __shfl(sIdx, m, 32);
            const int d = __shfl(dIdx, m, 32);
            const float4 a4 = ((const float4*)(x + (size_t)s * HDIM))[lane];
            const float4 c4 = ((const float4*)(x + (size_t)d * HDIM))[lane];
            v4h hv = { (f16)(a4.x * c4.x), (f16)(a4.y * c4.y),
                       (f16)(a4.z * c4.z), (f16)(a4.w * c4.w) };
            *(v4h*)(Abuf + m * HDIM + lane * 4) = hv;
        }
        asm volatile("s_wait_dscnt 0x0" ::: "memory");

        // ---- Layer 1 (transposed): h1^T = W1^T @ A^T, relu+bias -> Hbuf row-major.
        {
            v16h bF[2][4];                       // edge-activation B fragments, hoisted
#pragma unroll
            for (int g = 0; g < 2; ++g)
#pragma unroll
                for (int kt = 0; kt < 4; ++kt)
                    bF[g][kt] = load_B_act(Abuf + g * 16 * HDIM, lane, kt);

#pragma unroll
            for (int mt = 0; mt < 8; ++mt) {
                v8f acc0 = {}, acc1 = {};
#pragma unroll
                for (int kt = 0; kt < 4; ++kt) {
                    v16h aw = *(const v16h*)(W1a + (mt * 4 + kt) * TILE_HALVES + lane * 16);
                    acc0 = __builtin_amdgcn_wmma_f32_16x16x32_f16(
                        false, aw, false, bF[0][kt], (short)0, acc0, false, false);
                    acc1 = __builtin_amdgcn_wmma_f32_16x16x32_f16(
                        false, aw, false, bF[1][kt], (short)0, acc1, false, false);
                }
                const v8f bb = *(const v8f*)(b1s + mt * 16 + mBase);
                v8h h0, h1;
#pragma unroll
                for (int r = 0; r < 8; ++r) {
                    float v0 = acc0[r] + bb[r]; v0 = v0 > 0.f ? v0 : 0.f;
                    float v1 = acc1[r] + bb[r]; v1 = v1 > 0.f ? v1 : 0.f;
                    h0[r] = (f16)v0; h1[r] = (f16)v1;
                }
                // Lane holds 8 contiguous channels of one edge -> single b128 store.
                *(v8h*)(Hbuf + (n2)      * HDIM + mt * 16 + mBase) = h0;
                *(v8h*)(Hbuf + (16 + n2) * HDIM + mt * 16 + mBase) = h1;
            }
        }
        asm volatile("s_wait_dscnt 0x0" ::: "memory");

        // ---- Layer 2 (transposed) fused with W3 dot:
        //      out = relu(h1 @ W2 + b2) @ W3 + b3.
        float dotA = 0.f, dotB = 0.f;
        {
            v16h hF[2][4];
#pragma unroll
            for (int g = 0; g < 2; ++g)
#pragma unroll
                for (int kt = 0; kt < 4; ++kt)
                    hF[g][kt] = load_B_act(Hbuf + g * 16 * HDIM, lane, kt);

#pragma unroll
            for (int mt = 0; mt < 8; ++mt) {
                v8f acc0 = {}, acc1 = {};
#pragma unroll
                for (int kt = 0; kt < 4; ++kt) {
                    v16h aw = *(const v16h*)(W2a + (mt * 4 + kt) * TILE_HALVES + lane * 16);
                    acc0 = __builtin_amdgcn_wmma_f32_16x16x32_f16(
                        false, aw, false, hF[0][kt], (short)0, acc0, false, false);
                    acc1 = __builtin_amdgcn_wmma_f32_16x16x32_f16(
                        false, aw, false, hF[1][kt], (short)0, acc1, false, false);
                }
                const v8f bb = *(const v8f*)(b2s + mt * 16 + mBase);
                const v8f w3 = *(const v8f*)(W3s + mt * 16 + mBase);
#pragma unroll
                for (int r = 0; r < 8; ++r) {
                    float v0 = acc0[r] + bb[r]; v0 = v0 > 0.f ? v0 : 0.f;
                    float v1 = acc1[r] + bb[r]; v1 = v1 > 0.f ? v1 : 0.f;
                    dotA += v0 * w3[r];
                    dotB += v1 * w3[r];
                }
            }
        }

        // Each edge's dot is split across lane n and lane n+16: one xor-16 combine.
        dotA += __shfl_xor(dotA, 16, 32);
        dotB += __shfl_xor(dotB, 16, 32);

        if (lane < 16) {
            const int eA = e0 + n2;
            const int eB = e0 + 16 + n2;
            if (eA < nEdges) out[eA] = dotA + b3v;
            if (eB < nEdges) out[eB] = dotB + b3v;
        }
    }
}

extern "C" void kernel_launch(void* const* d_in, const int* in_sizes, int n_in,
                              void* d_out, int out_size, void* d_ws, size_t ws_size,
                              hipStream_t stream) {
    const float*     x   = (const float*)d_in[0];
    const long long* src = (const long long*)d_in[1];   // jax int64
    const long long* dst = (const long long*)d_in[2];
    const float*     W1  = (const float*)d_in[3];
    const float*     b1  = (const float*)d_in[4];
    const float*     W2  = (const float*)d_in[5];
    const float*     b2  = (const float*)d_in[6];
    const float*     W3  = (const float*)d_in[7];
    const float*     b3  = (const float*)d_in[8];
    float*           out = (float*)d_out;

    const int nEdges = in_sizes[1];
    const int nTiles = (nEdges + M_PER_WAVE - 1) / M_PER_WAVE;
    const int blocks = 640;                       // 2560 waves; 2 blocks/WGP by LDS
    const int totalWaves = blocks * WAVES_PER_BLOCK;

    EdgeDecoder_wmma_kernel<<<blocks, THREADS, LDS_BYTES, stream>>>(
        x, src, dst, W1, b1, W2, b2, W3, b3, out, nEdges, nTiles, totalWaves);
}
